// SWformer_28870770164497
// MI455X (gfx1250) — matrix operand order repaired and checked by reading, not compiled, not run on hardware
//
#include <hip/hip_runtime.h>
#include <cstdint>
#include <cstddef>

// ---------------------------------------------------------------------------
// SWformer forward on gfx1250 (MI455X).  All contractions go through
// V_WMMA_F32_16X16X32_F16 (spike operands are exact in f16, f32 accumulate).
// Elementwise LIF / neg-IF / BatchNorm stay in fp32 VALU.
// ---------------------------------------------------------------------------

typedef __attribute__((ext_vector_type(16))) _Float16 v16h;
typedef __attribute__((ext_vector_type(8)))  float    v8f;

union AFrag { v16h v; uint4 q[2]; };
union CFrag { v8f  v; float f[8]; };

// ---------------- WMMA GEMM ------------------------------------------------
// D = A(MxKpad, f16 row-major) * B^T   where Bp is (N x Kpad, f16 row-major)
// Wave computes a 16 x (16*NT) tile of D.
// Store:  D[(m>>mpiShift)*imgStride + n*Mpi + (m & (Mpi-1))] = acc + bias[n]
// (per-image (Cout, HW) layout -> lands directly in NCHW activations).
// Mpi is always a power of two (16/256/1024/4096) and tiles never straddle
// an image boundary, so the index math is shift/mask only.
// Prefetches are unconditional: gfx1250 prefetch is speculative (TH=0) and
// silently drops invalid addresses, so running one K-tile past the end of a
// workspace buffer is safe and keeps the inner loop branch-free.
template<int NT>
__global__ __launch_bounds__(32)
void wmma_gemm_k(const _Float16* __restrict__ A, const _Float16* __restrict__ Bp,
                 const float* __restrict__ bias, float* __restrict__ D,
                 int Kpad, int mpiShift, long imgStride)
{
    const int m0   = blockIdx.x << 4;
    const int n0   = blockIdx.y * (NT << 4);
    const int lane = threadIdx.x;
    const int r    = lane & 15;
    const int hi   = lane >> 4;

    const _Float16* Arow = A + (long)(m0 + r) * Kpad;

    CFrag acc[NT];
#pragma unroll
    for (int nt = 0; nt < NT; ++nt)
#pragma unroll
        for (int i = 0; i < 8; ++i) acc[nt].f[i] = 0.f;

    for (int k0 = 0; k0 < Kpad; k0 += 32) {
        // A fragment: lane covers K = {hi*8 .. hi*8+7} and {16+hi*8 ..} -> 2 x b128
        AFrag a;
        a.q[0] = *(const uint4*)(Arow + k0 + (hi << 3));
        a.q[1] = *(const uint4*)(Arow + k0 + 16 + (hi << 3));
        __builtin_prefetch(Arow + k0 + 32, 0, 1);   // global_prefetch_b8 (speculative)
#pragma unroll
        for (int nt = 0; nt < NT; ++nt) {
            const _Float16* Brow = Bp + (long)(n0 + (nt << 4) + r) * Kpad;
            // B fragment: lane covers K = {hi*8 .. hi*8+7} and {16+hi*8 ..} -> 2 x b128
            AFrag b;
            b.q[0] = *(const uint4*)(Brow + k0 + (hi << 3));
            b.q[1] = *(const uint4*)(Brow + k0 + 16 + (hi << 3));
            __builtin_prefetch(Brow + k0 + 32, 0, 1);
            acc[nt].v = __builtin_amdgcn_wmma_f32_16x16x32_f16(
                false, a.v, false, b.v, (short)0, acc[nt].v, false, false);
        }
    }

    const int Mpi = 1 << mpiShift;
    const int ml0 = m0 & (Mpi - 1);
    float* Dimg = D + (long)(m0 >> mpiShift) * imgStride;
#pragma unroll
    for (int nt = 0; nt < NT; ++nt) {
        const int n = n0 + (nt << 4) + r;
        const float bv = bias ? bias[n] : 0.f;
        float* Dp = Dimg + (long)n * Mpi + ml0 + (hi << 3);
#pragma unroll
        for (int v = 0; v < 8; ++v)               // C layout: row = v + 8*hi
            Dp[v] = acc[nt].f[v] + bv;
    }
}

// ---------------- im2col (f32 -> f16, K zero-padded) -----------------------
__global__ void im2col_k(const float* __restrict__ src, _Float16* __restrict__ dst,
                         long total, int imgStride, int c0, int Cin,
                         int H, int W, int KH, int Kpad)
{
    long idx = (long)blockIdx.x * 256 + threadIdx.x;
    if (idx >= total) return;
    int  k  = (int)(idx % Kpad);
    long m  = idx / Kpad;
    int  hw = (int)(m % (long)(H * W));
    int  im = (int)(m / (long)(H * W));
    int  oh = hw / W, ow = hw % W;
    int  KK = KH * KH;
    float val = 0.f;
    if (k < Cin * KK) {
        int ci = k / KK, rr = k % KK;
        int kh = rr / KH, kw = rr % KH;
        int pad = KH >> 1;
        int ih = oh + kh - pad, iw = ow + kw - pad;
        if (ih >= 0 && ih < H && iw >= 0 && iw < W)
            val = src[(long)im * imgStride + (long)(c0 + ci) * H * W + ih * W + iw];
    }
    dst[idx] = (_Float16)val;
}

// pack (R x Ks) fp32 matrix (optionally transposed) into (R x Kpad) f16
__global__ void pack_mat_k(const float* __restrict__ src, _Float16* __restrict__ dst,
                           int R, int Ks, int Kpad, int trans)
{
    int idx = blockIdx.x * 256 + threadIdx.x;
    if (idx >= R * Kpad) return;
    int k = idx % Kpad, r = idx / Kpad;
    float v = 0.f;
    if (k < Ks) v = trans ? src[k * R + r] : src[r * Ks + k];
    dst[idx] = (_Float16)v;
}

// per-16x16-tile transpose + cast: dst[t][a][b(pad32)] = src[t][b][a]
__global__ void pack_tileT_k(const float* __restrict__ src, _Float16* __restrict__ dst,
                             long total)
{
    long idx = (long)blockIdx.x * 256 + threadIdx.x;
    if (idx >= total) return;
    int b = (int)(idx & 31);
    int a = (int)((idx >> 5) & 15);
    long t = idx >> 9;
    dst[idx] = (b < 16) ? (_Float16)src[t * 256 + b * 16 + a] : (_Float16)0.f;
}

// ---------------- BatchNorm (training mode) --------------------------------
__global__ void bn_stats_k(const float* __restrict__ x, float* __restrict__ mean,
                           float* __restrict__ rstd, int N, int C, int HW)
{
    int c = blockIdx.x;
    __shared__ float ssum[256], ssq[256];
    float s = 0.f, q = 0.f;
    for (int n = 0; n < N; ++n) {
        const float* p = x + ((long)n * C + c) * HW;
        for (int i = threadIdx.x; i < HW; i += 256) { float v = p[i]; s += v; q += v * v; }
    }
    ssum[threadIdx.x] = s; ssq[threadIdx.x] = q;
    __syncthreads();
    for (int st = 128; st > 0; st >>= 1) {
        if (threadIdx.x < st) { ssum[threadIdx.x] += ssum[threadIdx.x + st];
                                ssq[threadIdx.x]  += ssq[threadIdx.x + st]; }
        __syncthreads();
    }
    if (threadIdx.x == 0) {
        float inv = 1.f / (float)(N * HW);
        float m = ssum[0] * inv;
        float var = ssq[0] * inv - m * m;
        mean[c] = m;
        rstd[c] = rsqrtf(var + 1e-5f);
    }
}

__global__ void bn_apply_k(const float* __restrict__ x, const float* __restrict__ res,
                           float* __restrict__ y, const float* __restrict__ mean,
                           const float* __restrict__ rstd, const float* __restrict__ g,
                           const float* __restrict__ b, int C, int HW, long total)
{
    long i = (long)blockIdx.x * 256 + threadIdx.x;
    if (i >= total) return;
    int c = (int)((i / HW) % C);
    float v = (x[i] - mean[c]) * rstd[c] * g[c] + b[c];
    if (res) v += res[i];
    y[i] = v;
}

// ---------------- spiking neurons (T = 4) ----------------------------------
__global__ void lif4_k(const float* __restrict__ x, float* __restrict__ s, int M)
{
    int i = blockIdx.x * 256 + threadIdx.x;
    if (i >= M) return;
    float v = 0.f;
#pragma unroll
    for (int t = 0; t < 4; ++t) {
        float xt = x[(long)t * M + i];
        v += (xt - v) * 0.5f;                // tau = 2
        float sp = (v >= 1.f) ? 1.f : 0.f;   // spike(v - 1)
        s[(long)t * M + i] = sp;
        v *= (1.f - sp);                     // hard reset
    }
}

__global__ void negif4_k(const float* __restrict__ x, float* __restrict__ s, int M)
{
    int i = blockIdx.x * 256 + threadIdx.x;
    if (i >= M) return;
    float v = 0.f;
#pragma unroll
    for (int t = 0; t < 4; ++t) {
        v += x[(long)t * M + i];
        float sp = ((v >= 1.f) ? 1.f : 0.f) - ((v <= -1.f) ? 1.f : 0.f);
        s[(long)t * M + i] = sp;
        v *= (1.f - fabsf(sp));
    }
}

// ---------------- maxpool 3x3 stride 2 pad 1 -------------------------------
__global__ void maxpool_k(const float* __restrict__ x, float* __restrict__ y,
                          int NC, int H, int W)
{
    int Ho = H >> 1, Wo = W >> 1;
    long total = (long)NC * Ho * Wo;
    long idx = (long)blockIdx.x * 256 + threadIdx.x;
    if (idx >= total) return;
    int wo = (int)(idx % Wo);
    int ho = (int)((idx / Wo) % Ho);
    long nc = idx / ((long)Wo * Ho);
    const float* p = x + nc * H * W;
    float m = -3.4e38f;
    for (int dh = -1; dh <= 1; ++dh) {
        int h = 2 * ho + dh; if (h < 0 || h >= H) continue;
        for (int dw = -1; dw <= 1; ++dw) {
            int w = 2 * wo + dw; if (w < 0 || w >= W) continue;
            m = fmaxf(m, p[h * W + w]);
        }
    }
    y[idx] = m;
}

__global__ void add2_k(float* __restrict__ dst, const float* __restrict__ src, int n)
{
    int i = blockIdx.x * 256 + threadIdx.x;
    if (i < n) dst[i] += src[i];
}

// ---------------- Haar matrix (orthonormal, n = 16) ------------------------
__global__ void make_haar_k(float* Hm)
{
    int i = threadIdx.x;                 // 256 entries
    int r = i >> 4, c = i & 15;
    float v;
    if (r == 0) v = 0.25f;
    else {
        int p = 31 - __clz(r);
        int q = r - (1 << p);
        int blk = 16 >> p;
        int st = q * blk;
        float s = sqrtf((float)(1 << p)) * 0.25f;
        if (c >= st && c < st + (blk >> 1))            v = s;
        else if (c >= st + (blk >> 1) && c < st + blk) v = -s;
        else                                           v = 0.f;
    }
    Hm[i] = v;
}

// ---------------- head -----------------------------------------------------
__global__ void reduce_feat_k(const float* __restrict__ y, float* __restrict__ feat)
{
    int idx = blockIdx.x * 256 + threadIdx.x;   // 8*256
    if (idx >= 8 * 256) return;
    int b = idx >> 8, c = idx & 255;
    float s = 0.f;
    for (int t = 0; t < 4; ++t) {
        const float* p = y + ((long)(t * 8 + b) * 256 + c) * 256;
        for (int i = 0; i < 256; ++i) s += p[i];
    }
    feat[idx] = s * (1.f / 1024.f);             // mean over HW and T
}

__global__ void head_k(const float* __restrict__ feat, const float* __restrict__ w,
                       const float* __restrict__ hb, float* __restrict__ out)
{
    int idx = blockIdx.x * 128 + threadIdx.x;
    if (idx >= 88) return;
    int b = idx / 11, k = idx % 11;
    float s = hb[k];
    for (int c = 0; c < 256; ++c) s += feat[b * 256 + c] * w[k * 256 + c];
    out[idx] = s;
}

// ===========================================================================
extern "C" void kernel_launch(void* const* d_in, const int* in_sizes, int n_in,
                              void* d_out, int out_size, void* d_ws, size_t ws_size,
                              hipStream_t stream)
{
    (void)in_sizes; (void)n_in; (void)out_size; (void)ws_size;
    auto P = [&](int i) { return (const float*)d_in[i]; };
    const float* X = P(0);
    const float *w0=P(1),*g0=P(2),*b0=P(3),*w1=P(4),*g1=P(5),*b1=P(6),
                *w2=P(7),*g2=P(8),*b2=P(9),*w3=P(10),*g3=P(11),*b3=P(12),
                *wr=P(13),*gr=P(14),*br=P(15);
    const float *headw = P(62), *headb = P(63);

    // ---- workspace bump allocator (deterministic layout every call) ----
    uintptr_t cur = (uintptr_t)d_ws;
    auto alloc = [&](size_t bytes) -> void* {
        uintptr_t p = (cur + 255) & ~(uintptr_t)255; cur = p + bytes; return (void*)p;
    };
    const long E16 = 2097152;            // (32,256,256) fp32 activation
    float* Y     = (float*)alloc(E16 * 4);
    float* XFEAT = (float*)alloc(E16 * 4);
    float* XS    = (float*)alloc(E16 * 4);
    float* T1    = (float*)alloc(E16 * 4);
    float* T2    = (float*)alloc(E16 * 4);
    float* SA    = (float*)alloc(16777216ull * 4);
    float* SB    = (float*)alloc(16777216ull * 4);
    float* HM    = (float*)alloc(256 * 4);
    float* MEAN  = (float*)alloc(1024 * 4);
    float* RSTD  = (float*)alloc(1024 * 4);
    float* FEAT  = (float*)alloc(2048 * 4);
    _Float16* ABUF = (_Float16*)alloc(8388608ull * 2);
    _Float16* WC0  = (_Float16*)alloc(32   * 32   * 2);
    _Float16* WC1  = (_Float16*)alloc(64   * 288  * 2);
    _Float16* WC2  = (_Float16*)alloc(128  * 576  * 2);
    _Float16* WC3  = (_Float16*)alloc(256  * 1152 * 2);
    _Float16* WCR  = (_Float16*)alloc(256  * 2304 * 2);
    _Float16* WFC1 = (_Float16*)alloc(1024 * 256  * 2);
    _Float16* WFC2 = (_Float16*)alloc(256  * 1024 * 2);
    _Float16* WG1  = (_Float16*)alloc(16 * 32  * 2);
    _Float16* WG2  = (_Float16*)alloc(16 * 160 * 2);
    _Float16* WMIX = (_Float16*)alloc(16 * 32  * 2);
    _Float16* HMP  = (_Float16*)alloc(16 * 32  * 2);
    _Float16* HMTP = (_Float16*)alloc(16 * 32  * 2);

    auto GEMM = [&](const _Float16* A, const _Float16* Bp, const float* bias, float* D,
                    int M, int N, int K, int Mpi, long imgStride) {
        int shift = __builtin_ctz((unsigned)Mpi);       // Mpi in {16,256,1024,4096}
        if ((N & 63) == 0) {
            dim3 g(M / 16, N / 64);
            wmma_gemm_k<4><<<g, 32, 0, stream>>>(A, Bp, bias, D, K, shift, imgStride);
        } else {
            dim3 g(M / 16, N / 16);
            wmma_gemm_k<1><<<g, 32, 0, stream>>>(A, Bp, bias, D, K, shift, imgStride);
        }
    };
    auto IM2COL = [&](const float* src, int Nimg, int imgStride, int c0, int Cin,
                      int H, int W, int KH, int Kpad) {
        long tot = (long)Nimg * H * W * Kpad;
        im2col_k<<<(tot + 255) / 256, 256, 0, stream>>>(src, ABUF, tot, imgStride,
                                                        c0, Cin, H, W, KH, Kpad);
    };
    auto PACK = [&](const float* src, _Float16* dst, int R, int Ks, int Kpad, int tr) {
        int tot = R * Kpad;
        pack_mat_k<<<(tot + 255) / 256, 256, 0, stream>>>(src, dst, R, Ks, Kpad, tr);
    };
    auto BN = [&](float* buf, const float* g, const float* b, int C, int HW) {
        bn_stats_k<<<C, 256, 0, stream>>>(buf, MEAN, RSTD, 32, C, HW);
        long tot = (long)32 * C * HW;
        bn_apply_k<<<(tot + 255) / 256, 256, 0, stream>>>(buf, nullptr, buf, MEAN, RSTD,
                                                          g, b, C, HW, tot);
    };
    auto LIF = [&](const float* in, float* out, int M) {
        lif4_k<<<(M + 255) / 256, 256, 0, stream>>>(in, out, M);
    };
    auto NEGIF = [&](const float* in, float* out, int M) {
        negif4_k<<<(M + 255) / 256, 256, 0, stream>>>(in, out, M);
    };
    auto ADD = [&](float* dst, const float* src, int n) {
        add2_k<<<(n + 255) / 256, 256, 0, stream>>>(dst, src, n);
    };
    auto TILET = [&](const float* src) {     // 8192 tiles of 16x16 -> (N x 32) f16
        long tot = 8192L * 512;
        pack_tileT_k<<<(tot + 255) / 256, 256, 0, stream>>>(src, ABUF, tot);
    };

    // ---- constants / packed weights ----
    make_haar_k<<<1, 256, 0, stream>>>(HM);
    PACK(HM, HMP, 16, 16, 32, 0);
    PACK(HM, HMTP, 16, 16, 32, 1);
    PACK(w0, WC0, 32, 18, 32, 0);
    PACK(w1, WC1, 64, 288, 288, 0);
    PACK(w2, WC2, 128, 576, 576, 0);
    PACK(w3, WC3, 256, 1152, 1152, 0);
    PACK(wr, WCR, 256, 2304, 2304, 0);

    // ================= SPS patch embedding =================
    // conv0 (whole batch): (32img, 2ch, 64x64) -> (32, 32, 64x64)
    IM2COL(X, 32, 2 * 4096, 0, 2, 64, 64, 3, 32);
    GEMM(ABUF, WC0, nullptr, SA, 131072, 32, 32, 4096, 32L * 4096);
    BN(SA, g0, b0, 32, 4096);
    LIF(SA, SB, 8 * 32 * 4096);
    // conv1 per image
    for (int i = 0; i < 32; ++i) {
        IM2COL(SB + (long)i * 32 * 4096, 1, 0, 0, 32, 64, 64, 3, 288);
        GEMM(ABUF, WC1, nullptr, SA + (long)i * 64 * 4096, 4096, 64, 288, 4096, 0);
    }
    BN(SA, g1, b1, 64, 4096);
    LIF(SA, SB, 8 * 64 * 4096);
    // conv2 per image
    for (int i = 0; i < 32; ++i) {
        IM2COL(SB + (long)i * 64 * 4096, 1, 0, 0, 64, 64, 64, 3, 576);
        GEMM(ABUF, WC2, nullptr, SA + (long)i * 128 * 4096, 4096, 128, 576, 4096, 0);
    }
    BN(SA, g2, b2, 128, 4096);
    LIF(SA, SB, 8 * 128 * 4096);
    maxpool_k<<<(4194304 + 255) / 256, 256, 0, stream>>>(SB, SA, 32 * 128, 64, 64);
    // conv3 per image at 32x32
    for (int i = 0; i < 32; ++i) {
        IM2COL(SA + (long)i * 128 * 1024, 1, 0, 0, 128, 32, 32, 3, 1152);
        GEMM(ABUF, WC3, nullptr, SB + (long)i * 256 * 1024, 1024, 256, 1152, 1024, 0);
    }
    BN(SB, g3, b3, 256, 1024);
    maxpool_k<<<(2097152 + 255) / 256, 256, 0, stream>>>(SB, XFEAT, 32 * 256, 32, 32);
    // residual conv at 16x16
    LIF(XFEAT, T1, 8 * 256 * 256);
    for (int i = 0; i < 32; ++i) {
        IM2COL(T1 + (long)i * 65536, 1, 0, 0, 256, 16, 16, 3, 2304);
        GEMM(ABUF, WCR, nullptr, T2 + (long)i * 65536, 256, 256, 2304, 256, 0);
    }
    bn_stats_k<<<256, 256, 0, stream>>>(T2, MEAN, RSTD, 32, 256, 256);
    bn_apply_k<<<(E16 + 255) / 256, 256, 0, stream>>>(T2, XFEAT, Y, MEAN, RSTD,
                                                      gr, br, 256, 256, E16);

    // ================= transformer blocks =================
    for (int d = 0; d < 2; ++d) {
        int base = 16 + d * 23;
        const float *hwt=P(base+0), *c1w=P(base+1), *c1b=P(base+2), *c2w=P(base+3),
                    *c2b=P(base+4), *hfg=P(base+5), *hfb=P(base+6), *hmg=P(base+7),
                    *hmb=P(base+8), *hig=P(base+9), *hib=P(base+10),
                    *c1g=P(base+11), *c1bb=P(base+12), *c2g=P(base+13), *c2bb=P(base+14),
                    *fc1w=P(base+15), *fc1cb=P(base+16), *fc1g=P(base+17), *fc1bb=P(base+18),
                    *fc2w=P(base+19), *fc2cb=P(base+20), *fc2g=P(base+21), *fc2bb=P(base+22);
        PACK(fc1w, WFC1, 1024, 256, 256, 0);
        PACK(fc2w, WFC2, 256, 1024, 1024, 0);
        PACK(c1w, WG1, 16, 16, 32, 0);
        PACK(c2w, WG2, 16, 144, 160, 0);

        // ---- FATM ----
        LIF(Y, XS, 524288);
        // forward Haar: Hm @ X @ Hm^T  (two M=16 GEMMs, N = 8192 tiles * 16)
        TILET(XS);
        GEMM(HMP, ABUF, nullptr, T1, 16, 131072, 32, 16, 0);
        TILET(T1);
        GEMM(HMP, ABUF, nullptr, T2, 16, 131072, 32, 16, 0);
        NEGIF(T2, T1, 524288);
        BN(T1, hfg, hfb, 256, 256);
        LIF(T1, T2, 524288);
        // block-diagonal mixing per 16-channel group
        for (int g = 0; g < 16; ++g) {
            IM2COL(T2, 32, 65536, g * 16, 16, 16, 16, 1, 32);
            PACK(hwt + g * 256, WMIX, 16, 16, 32, 1);
            GEMM(ABUF, WMIX, nullptr, T1 + (long)g * 16 * 256, 8192, 16, 32, 256, 65536);
        }
        BN(T1, hmg, hmb, 256, 256);
        NEGIF(T1, T2, 524288);
        // inverse Haar: Hm^T @ S @ Hm
        TILET(T2);
        GEMM(HMTP, ABUF, nullptr, T1, 16, 131072, 32, 16, 0);
        TILET(T1);
        GEMM(HMTP, ABUF, nullptr, T2, 16, 131072, 32, 16, 0);
        BN(T2, hig, hib, 256, 256);                  // h in T2
        // grouped 1x1 conv branch
        for (int g = 0; g < 16; ++g) {
            IM2COL(XS, 32, 65536, g * 16, 16, 16, 16, 1, 32);
            GEMM(ABUF, WG1, c1b, T1 + (long)g * 16 * 256, 8192, 16, 32, 256, 65536);
        }
        BN(T1, c1g, c1bb, 256, 256);
        ADD(T2, T1, (int)E16);
        // grouped 3x3 conv branch
        for (int g = 0; g < 16; ++g) {
            IM2COL(XS, 32, 65536, g * 16, 16, 16, 16, 3, 160);
            GEMM(ABUF, WG2, c2b, T1 + (long)g * 16 * 256, 8192, 16, 160, 256, 65536);
        }
        BN(T1, c2g, c2bb, 256, 256);
        ADD(T2, T1, (int)E16);
        ADD(Y, T2, (int)E16);                        // y = y + fatm(y)

        // ---- MLP ----
        LIF(Y, XS, 524288);
        IM2COL(XS, 32, 65536, 0, 256, 16, 16, 1, 256);
        GEMM(ABUF, WFC1, fc1cb, SA, 8192, 1024, 256, 256, 262144);
        BN(SA, fc1g, fc1bb, 1024, 256);
        LIF(SA, SB, 2097152);
        IM2COL(SB, 32, 262144, 0, 1024, 16, 16, 1, 1024);
        GEMM(ABUF, WFC2, fc2cb, T1, 8192, 256, 1024, 256, 65536);
        BN(T1, fc2g, fc2bb, 256, 256);
        ADD(Y, T1, (int)E16);                        // y = y + mlp(y)
    }

    // ================= head =================
    reduce_feat_k<<<8, 256, 0, stream>>>(Y, FEAT);
    head_k<<<1, 128, 0, stream>>>(FEAT, headw, headb, (float*)d_out);
}
